// EISA_18373870092742
// MI455X (gfx1250) — compile-verified, matrix operand order checked by reading, not run on hardware
//
#include <hip/hip_runtime.h>
#include <math.h>

typedef float     v4f  __attribute__((ext_vector_type(4)));
typedef float     v8f  __attribute__((ext_vector_type(8)));
typedef _Float16  v8h  __attribute__((ext_vector_type(8)));
typedef _Float16  v16h __attribute__((ext_vector_type(16)));
typedef unsigned int u32x4 __attribute__((ext_vector_type(4)));
typedef int          i32x4 __attribute__((ext_vector_type(4)));
typedef int          i32x8 __attribute__((ext_vector_type(8)));

#define BATCH 512
#define HH 7
#define WW 7
#define CC 768
#define MROWS 25088          // 512*7*7
#define KMIX 2304            // 3*768
#define BCHUNK 128           // K-halves per LDS-staged B chunk
#define BROWS 64             // N rows per block tile

#if defined(__has_builtin)
#if __has_builtin(__builtin_amdgcn_tensor_load_to_lds)
#define HAVE_TDM 1
#endif
#endif
#ifndef HAVE_TDM
#define HAVE_TDM 0
#endif

// ---------------------------------------------------------------------------
// helpers
// ---------------------------------------------------------------------------
__device__ __forceinline__ v8f wmma16x16x32(v16h a, v16h b, v8f c) {
    return __builtin_amdgcn_wmma_f32_16x16x32_f16(
        false, a, false, b, (short)0, c, false, false);
}

// f16 fragment from row-major [row][K] memory (global or LDS). p -> (row, kk+kof).
__device__ __forceinline__ v16h load_frag_f16(const _Float16* p) {
    v8h lo = *(const v8h*)(p);
    v8h hi = *(const v8h*)(p + 16);
    return __builtin_shufflevector(lo, hi, 0,1,2,3,4,5,6,7,8,9,10,11,12,13,14,15);
}

__device__ __forceinline__ v16h load_frag_f16_scaled(const _Float16* p, float g) {
    v8h lo = *(const v8h*)(p);
    v8h hi = *(const v8h*)(p + 16);
    _Float16 gh = (_Float16)g;
    v16h a;
#pragma unroll
    for (int i = 0; i < 8; ++i) { a[i] = lo[i] * gh; a[8 + i] = hi[i] * gh; }
    return a;
}

__device__ __forceinline__ v16h load_frag_f16_vscaled(const _Float16* p, const float* gp) {
    v8h lo = *(const v8h*)(p);
    v8h hi = *(const v8h*)(p + 16);
    v4f g0 = *(const v4f*)(gp);
    v4f g1 = *(const v4f*)(gp + 4);
    v4f g2 = *(const v4f*)(gp + 16);
    v4f g3 = *(const v4f*)(gp + 20);
    v16h a;
#pragma unroll
    for (int i = 0; i < 4; ++i) {
        a[i]      = (_Float16)((float)lo[i]     * g0[i]);
        a[4 + i]  = (_Float16)((float)lo[4 + i] * g1[i]);
        a[8 + i]  = (_Float16)((float)hi[i]     * g2[i]);
        a[12 + i] = (_Float16)((float)hi[4 + i] * g3[i]);
    }
    return a;
}

__device__ __forceinline__ v16h load_frag_f32(const float* p) {
    v4f f0 = *(const v4f*)(p);
    v4f f1 = *(const v4f*)(p + 4);
    v4f f2 = *(const v4f*)(p + 16);
    v4f f3 = *(const v4f*)(p + 20);
    v16h a;
#pragma unroll
    for (int i = 0; i < 4; ++i) {
        a[i]      = (_Float16)f0[i];
        a[4 + i]  = (_Float16)f1[i];
        a[8 + i]  = (_Float16)f2[i];
        a[12 + i] = (_Float16)f3[i];
    }
    return a;
}

__device__ __forceinline__ float sigmoidf_(float x) { return 1.0f / (1.0f + expf(-x)); }
__device__ __forceinline__ float gelu_exact(float v) {
    return 0.5f * v * (1.0f + erff(v * 0.70710678118654752f));
}

// ---------------------------------------------------------------------------
// B-tile staging: 64 rows x 128 halves from row-major weight (row stride K)
// into LDS. TDM path builds a 2-D D# per CDNA5 ISA.
// ---------------------------------------------------------------------------
__device__ __forceinline__ void stage_b_chunk(const _Float16* gsrc, _Float16* lds_dst,
                                              int K, int wave, int tid) {
#if HAVE_TDM
    if (wave == 0) {
        const unsigned long long ga = (unsigned long long)(uintptr_t)gsrc;
        const unsigned int lds_addr = (unsigned int)(uintptr_t)lds_dst;  // low 32 = LDS offset
        u32x4 g0;
        g0[0] = 1u;                                            // count=1 valid descriptor
        g0[1] = lds_addr;                                      // lds_addr [63:32]
        g0[2] = (unsigned int)(ga & 0xFFFFFFFFu);              // global_addr[31:0]
        g0[3] = (unsigned int)((ga >> 32) & 0x1FFFFFFu) | (2u << 30);  // ga[56:32] | type=2
        const unsigned int td0 = (unsigned int)K;              // tensor_dim0 (elements)
        const unsigned int td1 = (unsigned int)BROWS;          // tensor_dim1
        i32x8 g1;
        g1[0] = (int)(1u << 16);                               // data_size=1 (2 bytes)
        g1[1] = (int)((td0 & 0xFFFFu) << 16);                  // tensor_dim0[15:0]
        g1[2] = (int)(((td0 >> 16) & 0xFFFFu) | ((td1 & 0xFFFFu) << 16));
        g1[3] = (int)(((td1 >> 16) & 0xFFFFu) | ((unsigned)BCHUNK << 16)); // tile_dim0=128
        g1[4] = (int)((unsigned)BROWS);                        // tile_dim1=64, tile_dim2=0
        g1[5] = (int)(unsigned int)K;                          // tensor_dim0_stride[31:0]
        g1[6] = 0;                                             // stride[47:32], dim1_stride lo
        g1[7] = 0;
        i32x4 z4 = {0, 0, 0, 0};
        i32x8 z8 = {0, 0, 0, 0, 0, 0, 0, 0};
        __builtin_amdgcn_tensor_load_to_lds(g0, g1, z4, z4, z8, 0);
    }
#else
    // cooperative fallback: 64*128 halves = 1024 v8h pieces over 256 threads
    for (int i = tid; i < BROWS * BCHUNK / 8; i += 256) {
        const int row = i >> 4;            // 16 v8h per row
        const int col = (i & 15) * 8;
        *(v8h*)(lds_dst + row * BCHUNK + col) = *(const v8h*)(gsrc + (size_t)row * K + col);
    }
#endif
}

__device__ __forceinline__ void chunk_wait(int wave) {
#if HAVE_TDM
    if (wave == 0) __builtin_amdgcn_s_wait_tensorcnt(0);
#endif
    __syncthreads();
}

// ---------------------------------------------------------------------------
// generic fp32 -> f16 convert (weights)
// ---------------------------------------------------------------------------
__global__ void k_cvt_f16(const float* __restrict__ in, _Float16* __restrict__ out, int n) {
    int i = blockIdx.x * blockDim.x + threadIdx.x;
    if (i < n) out[i] = (_Float16)in[i];
}

// ---------------------------------------------------------------------------
// stage-1 GEMM: Out = f16( gelu( x(f32) @ W^T + bias ) )   M=25088, N=K=768
// ---------------------------------------------------------------------------
__global__ void k_gemm_s1(const float* __restrict__ A, const _Float16* __restrict__ W,
                          const float* __restrict__ bias, _Float16* __restrict__ Out) {
    __shared__ _Float16 lds_b[2][BROWS * BCHUNK];
    const int tid  = threadIdx.x;
    const int lane = tid & 31;
    const int wave = tid >> 5;
    const int l16  = lane & 15;
    const int kof  = (lane & 16) ? 8 : 0;
    const int m0   = blockIdx.x * 256 + wave * 32;
    const int n0   = blockIdx.y * 64;

    const float* ap[2];
#pragma unroll
    for (int mi = 0; mi < 2; ++mi) ap[mi] = A + (size_t)(m0 + mi * 16 + l16) * CC;
    const _Float16* wtile = W + (size_t)n0 * CC;

    v8f acc[2][4] = {};
    const int nchunks = CC / BCHUNK;            // 6
    stage_b_chunk(wtile, lds_b[0], CC, wave, tid);

#pragma unroll 1
    for (int c = 0; c < nchunks; ++c) {
        chunk_wait(wave);
        if (c + 1 < nchunks)
            stage_b_chunk(wtile + (c + 1) * BCHUNK, lds_b[(c + 1) & 1], CC, wave, tid);
        const _Float16* bb = lds_b[c & 1];
        const int kk0 = c * BCHUNK;
#pragma unroll
        for (int mi = 0; mi < 2; ++mi)
            __builtin_prefetch(ap[mi] + kk0 + BCHUNK + kof, 0, 1);
#pragma unroll
        for (int kk2 = 0; kk2 < BCHUNK; kk2 += 32) {
            v16h af[2], bf[4];
#pragma unroll
            for (int mi = 0; mi < 2; ++mi) af[mi] = load_frag_f32(ap[mi] + kk0 + kk2 + kof);
#pragma unroll
            for (int ni = 0; ni < 4; ++ni)
                bf[ni] = load_frag_f16(bb + (size_t)(ni * 16 + l16) * BCHUNK + kk2 + kof);
#pragma unroll
            for (int mi = 0; mi < 2; ++mi)
#pragma unroll
                for (int ni = 0; ni < 4; ++ni)
                    acc[mi][ni] = wmma16x16x32(af[mi], bf[ni], acc[mi][ni]);
        }
    }

    const int rowoff = (lane & 16) ? 8 : 0;
#pragma unroll
    for (int mi = 0; mi < 2; ++mi)
#pragma unroll
        for (int ni = 0; ni < 4; ++ni) {
            const int col  = n0 + ni * 16 + l16;
            const float bc = bias[col];
            const int rb   = m0 + mi * 16 + rowoff;
#pragma unroll
            for (int j = 0; j < 8; ++j) {
                float v = acc[mi][ni][j] + bc;
                Out[(size_t)(rb + j) * CC + col] = (_Float16)gelu_exact(v);
            }
        }
}

// ---------------------------------------------------------------------------
// pooling kernels
// ---------------------------------------------------------------------------
__global__ void k_pool_h(const _Float16* __restrict__ h16, float* __restrict__ h2v) {
    __shared__ float red[256];
    const size_t base = (size_t)blockIdx.x * (WW * CC);
    float s = 0.f;
    for (int i = threadIdx.x; i < WW * CC; i += 256) s += (float)h16[base + i];
    red[threadIdx.x] = s; __syncthreads();
    for (int st = 128; st > 0; st >>= 1) {
        if (threadIdx.x < st) red[threadIdx.x] += red[threadIdx.x + st];
        __syncthreads();
    }
    if (threadIdx.x == 0) h2v[blockIdx.x] = red[0] * (1.0f / (WW * CC));
}

__global__ void k_pool_w(const _Float16* __restrict__ w16, float* __restrict__ w2v) {
    __shared__ float red[256];
    const int b = blockIdx.x / WW, widx = blockIdx.x % WW;
    float s = 0.f;
    for (int hidx = 0; hidx < HH; ++hidx) {
        const size_t base = ((size_t)(b * HH + hidx) * WW + widx) * CC;
        for (int c = threadIdx.x; c < CC; c += 256) s += (float)w16[base + c];
    }
    red[threadIdx.x] = s; __syncthreads();
    for (int st = 128; st > 0; st >>= 1) {
        if (threadIdx.x < st) red[threadIdx.x] += red[threadIdx.x + st];
        __syncthreads();
    }
    if (threadIdx.x == 0) w2v[blockIdx.x] = red[0] * (1.0f / (HH * CC));
}

__global__ void k_pool_s(const _Float16* __restrict__ s16, float* __restrict__ s1v) {
    const int b = blockIdx.x;
    for (int c = threadIdx.x; c < CC; c += 256) {
        const size_t base = (size_t)b * (HH * WW) * CC + c;
        float s = 0.f;
        for (int p = 0; p < HH * WW; ++p) s += (float)s16[base + (size_t)p * CC];
        s1v[(size_t)b * CC + c] = s * (1.0f / (HH * WW));
    }
}

// ---------------------------------------------------------------------------
// gates: per-batch sigmoid / conv3 / linear mixing
// ---------------------------------------------------------------------------
__global__ void k_gates(const float* __restrict__ h2v, const float* __restrict__ w2v,
                        const float* __restrict__ s1v,
                        const float* convH_w, const float* convH_b,
                        const float* convH2_k, const float* convH2_b,
                        const float* mixhw_k, const float* mixhw_b,
                        const float* mixhw2_k, const float* mixhw2_b,
                        const float* mixhws_k, const float* mixhws_b,
                        const float* mixhws2_k, const float* mixhws2_b,
                        const float* __restrict__ linH_W, const float* __restrict__ linH_b,
                        const float* __restrict__ linW_W, const float* __restrict__ linW_b,
                        float* __restrict__ g_h, float* __restrict__ g_hw,
                        float* __restrict__ g_c) {
    __shared__ float sh_h2[HH], sh_w2[HH], sh_th[HH], sh_thw[HH], sm[CC];
    const int b = blockIdx.x, tid = threadIdx.x;
    if (tid < HH) {
        float hv = h2v[b * HH + tid];
        float wv = w2v[b * HH + tid];
        sh_h2[tid] = hv; sh_w2[tid] = wv;
        sh_th[tid]  = sigmoidf_(hv * convH_w[0] + convH_b[0]);
        sh_thw[tid] = sigmoidf_(mixhw_k[0] * hv + mixhw_k[1] * wv + mixhw_b[0]);
    }
    __syncthreads();
    if (tid < HH) {
        float l = (tid > 0)      ? sh_th[tid - 1] : 0.f;
        float c = sh_th[tid];
        float r = (tid < HH - 1) ? sh_th[tid + 1] : 0.f;
        g_h[b * HH + tid] = sigmoidf_(convH2_k[0] * l + convH2_k[1] * c + convH2_k[2] * r + convH2_b[0]);
        l = (tid > 0)      ? sh_thw[tid - 1] : 0.f;
        c = sh_thw[tid];
        r = (tid < HH - 1) ? sh_thw[tid + 1] : 0.f;
        g_hw[b * HH + tid] = sigmoidf_(mixhw2_k[0] * l + mixhw2_k[1] * c + mixhw2_k[2] * r + mixhw2_b[0]);
    }
    for (int c = tid; c < CC; c += 256) {
        float hl = linH_b[c], wl = linW_b[c];
#pragma unroll
        for (int i = 0; i < HH; ++i) {
            hl += linH_W[c * HH + i] * sh_h2[i];
            wl += linW_W[c * HH + i] * sh_w2[i];
        }
        sm[c] = sigmoidf_(mixhws_k[0] * hl + mixhws_k[1] * wl +
                          mixhws_k[2] * s1v[(size_t)b * CC + c] + mixhws_b[0]);
    }
    __syncthreads();
    for (int c = tid; c < CC; c += 256) {
        float l = (c > 0)      ? sm[c - 1] : 0.f;
        float ce = sm[c];
        float r = (c < CC - 1) ? sm[c + 1] : 0.f;
        g_c[(size_t)b * CC + c] =
            sigmoidf_(mixhws2_k[0] * l + mixhws2_k[1] * ce + mixhws2_k[2] * r + mixhws2_b[0]);
    }
}

// ---------------------------------------------------------------------------
// mixall GEMM with fused gating of the A operand, B staged via TDM.
// ---------------------------------------------------------------------------
__global__ void k_gemm_mix(const _Float16* __restrict__ h16, const _Float16* __restrict__ w16,
                           const _Float16* __restrict__ s16, const _Float16* __restrict__ Wmix,
                           const float* __restrict__ bias, const float* __restrict__ g_h,
                           const float* __restrict__ g_hw, const float* __restrict__ g_c,
                           _Float16* __restrict__ Out) {
    __shared__ _Float16 lds_b[2][BROWS * BCHUNK];
    const int tid  = threadIdx.x;
    const int lane = tid & 31;
    const int wave = tid >> 5;
    const int l16  = lane & 15;
    const int kof  = (lane & 16) ? 8 : 0;
    const int m0   = blockIdx.x * 256 + wave * 32;
    const int n0   = blockIdx.y * 64;

    const _Float16* hp[2];
    const _Float16* wp[2];
    const _Float16* sp[2];
    const float*    gp[2];
    float gh[2], ghw[2];
#pragma unroll
    for (int mi = 0; mi < 2; ++mi) {
        const int r   = m0 + mi * 16 + l16;
        const int b   = r / (HH * WW);
        const int rem = r % (HH * WW);
        hp[mi]  = h16 + (size_t)r * CC;
        wp[mi]  = w16 + (size_t)r * CC;
        sp[mi]  = s16 + (size_t)r * CC;
        gp[mi]  = g_c + (size_t)b * CC;
        gh[mi]  = g_h[b * HH + rem / WW];
        ghw[mi] = g_hw[b * HH + rem % WW];
    }
    const _Float16* wtile = Wmix + (size_t)n0 * KMIX;

    v8f acc[2][4] = {};
    const int nchunks = KMIX / BCHUNK;          // 18
    stage_b_chunk(wtile, lds_b[0], KMIX, wave, tid);

#pragma unroll 1
    for (int c = 0; c < nchunks; ++c) {
        chunk_wait(wave);
        if (c + 1 < nchunks)
            stage_b_chunk(wtile + (c + 1) * BCHUNK, lds_b[(c + 1) & 1], KMIX, wave, tid);
        const _Float16* bb = lds_b[c & 1];
        const int kk0 = c * BCHUNK;
#pragma unroll
        for (int kk2 = 0; kk2 < BCHUNK; kk2 += 32) {
            const int kk = kk0 + kk2;
            v16h af[2], bf[4];
            if (kk < CC) {
#pragma unroll
                for (int mi = 0; mi < 2; ++mi)
                    af[mi] = load_frag_f16_scaled(hp[mi] + kk + kof, gh[mi]);
            } else if (kk < 2 * CC) {
#pragma unroll
                for (int mi = 0; mi < 2; ++mi)
                    af[mi] = load_frag_f16_scaled(wp[mi] + (kk - CC) + kof, ghw[mi]);
            } else {
                const int k0 = kk - 2 * CC;
#pragma unroll
                for (int mi = 0; mi < 2; ++mi)
                    af[mi] = load_frag_f16_vscaled(sp[mi] + k0 + kof, gp[mi] + k0 + kof);
            }
#pragma unroll
            for (int ni = 0; ni < 4; ++ni)
                bf[ni] = load_frag_f16(bb + (size_t)(ni * 16 + l16) * BCHUNK + kk2 + kof);
#pragma unroll
            for (int mi = 0; mi < 2; ++mi)
#pragma unroll
                for (int ni = 0; ni < 4; ++ni)
                    acc[mi][ni] = wmma16x16x32(af[mi], bf[ni], acc[mi][ni]);
        }
    }

    const int rowoff = (lane & 16) ? 8 : 0;
#pragma unroll
    for (int mi = 0; mi < 2; ++mi)
#pragma unroll
        for (int ni = 0; ni < 4; ++ni) {
            const int col  = n0 + ni * 16 + l16;
            const float bc = bias[col];
            const int rb   = m0 + mi * 16 + rowoff;
#pragma unroll
            for (int j = 0; j < 8; ++j)
                Out[(size_t)(rb + j) * CC + col] = (_Float16)(acc[mi][ni][j] + bc);
        }
}

// ---------------------------------------------------------------------------
// proj GEMM + residual: out = (mixed @ proj^T + proj_b)*gw0 + x*gw1  (f32 out)
// ---------------------------------------------------------------------------
__global__ void k_gemm_proj(const _Float16* __restrict__ A, const _Float16* __restrict__ W,
                            const float* __restrict__ bias, const float* __restrict__ xres,
                            const float* __restrict__ gatew, float* __restrict__ Out) {
    __shared__ _Float16 lds_b[2][BROWS * BCHUNK];
    const int tid  = threadIdx.x;
    const int lane = tid & 31;
    const int wave = tid >> 5;
    const int l16  = lane & 15;
    const int kof  = (lane & 16) ? 8 : 0;
    const int m0   = blockIdx.x * 256 + wave * 32;
    const int n0   = blockIdx.y * 64;

    const _Float16* ap[2];
#pragma unroll
    for (int mi = 0; mi < 2; ++mi) ap[mi] = A + (size_t)(m0 + mi * 16 + l16) * CC;
    const _Float16* wtile = W + (size_t)n0 * CC;

    v8f acc[2][4] = {};
    const int nchunks = CC / BCHUNK;            // 6
    stage_b_chunk(wtile, lds_b[0], CC, wave, tid);

#pragma unroll 1
    for (int c = 0; c < nchunks; ++c) {
        chunk_wait(wave);
        if (c + 1 < nchunks)
            stage_b_chunk(wtile + (c + 1) * BCHUNK, lds_b[(c + 1) & 1], CC, wave, tid);
        const _Float16* bb = lds_b[c & 1];
        const int kk0 = c * BCHUNK;
#pragma unroll
        for (int mi = 0; mi < 2; ++mi)
            __builtin_prefetch(ap[mi] + kk0 + BCHUNK + kof, 0, 1);
#pragma unroll
        for (int kk2 = 0; kk2 < BCHUNK; kk2 += 32) {
            v16h af[2], bf[4];
#pragma unroll
            for (int mi = 0; mi < 2; ++mi) af[mi] = load_frag_f16(ap[mi] + kk0 + kk2 + kof);
#pragma unroll
            for (int ni = 0; ni < 4; ++ni)
                bf[ni] = load_frag_f16(bb + (size_t)(ni * 16 + l16) * BCHUNK + kk2 + kof);
#pragma unroll
            for (int mi = 0; mi < 2; ++mi)
#pragma unroll
                for (int ni = 0; ni < 4; ++ni)
                    acc[mi][ni] = wmma16x16x32(af[mi], bf[ni], acc[mi][ni]);
        }
    }

    const float gw0 = gatew[0], gw1 = gatew[1];
    const int rowoff = (lane & 16) ? 8 : 0;
#pragma unroll
    for (int mi = 0; mi < 2; ++mi)
#pragma unroll
        for (int ni = 0; ni < 4; ++ni) {
            const int col  = n0 + ni * 16 + l16;
            const float bc = bias[col];
            const int rb   = m0 + mi * 16 + rowoff;
#pragma unroll
            for (int j = 0; j < 8; ++j) {
                const size_t idx = (size_t)(rb + j) * CC + col;
                Out[idx] = (acc[mi][ni][j] + bc) * gw0 + xres[idx] * gw1;
            }
        }
}

// ---------------------------------------------------------------------------
// host launch
// ---------------------------------------------------------------------------
extern "C" void kernel_launch(void* const* d_in, const int* in_sizes, int n_in,
                              void* d_out, int out_size, void* d_ws, size_t ws_size,
                              hipStream_t stream) {
    const float* x        = (const float*)d_in[0];
    const float* Wh       = (const float*)d_in[1];
    const float* bh       = (const float*)d_in[2];
    const float* Ww_      = (const float*)d_in[3];
    const float* bw       = (const float*)d_in[4];
    const float* Ws       = (const float*)d_in[5];
    const float* bs       = (const float*)d_in[6];
    const float* convH_w  = (const float*)d_in[7];
    const float* convH_b  = (const float*)d_in[8];
    const float* convH2_k = (const float*)d_in[9];
    const float* convH2_b = (const float*)d_in[10];
    const float* mixhw_k  = (const float*)d_in[11];
    const float* mixhw_b  = (const float*)d_in[12];
    const float* mixhw2_k = (const float*)d_in[13];
    const float* mixhw2_b = (const float*)d_in[14];
    const float* mixhws_k = (const float*)d_in[15];
    const float* mixhws_b = (const float*)d_in[16];
    const float* mixhws2_k= (const float*)d_in[17];
    const float* mixhws2_b= (const float*)d_in[18];
    const float* linH_W   = (const float*)d_in[19];
    const float* linH_b   = (const float*)d_in[20];
    const float* linW_W   = (const float*)d_in[21];
    const float* linW_b   = (const float*)d_in[22];
    const float* mixall_W = (const float*)d_in[23];
    const float* mixall_b = (const float*)d_in[24];
    const float* proj_W   = (const float*)d_in[25];
    const float* proj_b   = (const float*)d_in[26];
    const float* gate_w   = (const float*)d_in[27];

    float* out = (float*)d_out;

    char* base = (char*)d_ws;
    size_t off = 0;
    auto alloc = [&](size_t bytes) {
        void* p = base + off;
        off += (bytes + 255) & ~(size_t)255;
        return p;
    };
    const size_t nAct = (size_t)MROWS * CC;
    _Float16* h16    = (_Float16*)alloc(nAct * 2);
    _Float16* w16    = (_Float16*)alloc(nAct * 2);
    _Float16* s16    = (_Float16*)alloc(nAct * 2);
    _Float16* mix16  = (_Float16*)alloc(nAct * 2);
    _Float16* Wh16   = (_Float16*)alloc((size_t)CC * CC * 2);
    _Float16* Ww16   = (_Float16*)alloc((size_t)CC * CC * 2);
    _Float16* Ws16   = (_Float16*)alloc((size_t)CC * CC * 2);
    _Float16* Wp16   = (_Float16*)alloc((size_t)CC * CC * 2);
    _Float16* Wm16   = (_Float16*)alloc((size_t)CC * KMIX * 2);
    float* h2v       = (float*)alloc((size_t)BATCH * HH * 4);
    float* w2v       = (float*)alloc((size_t)BATCH * WW * 4);
    float* s1v       = (float*)alloc((size_t)BATCH * CC * 4);
    float* g_h       = (float*)alloc((size_t)BATCH * HH * 4);
    float* g_hw      = (float*)alloc((size_t)BATCH * WW * 4);
    float* g_c       = (float*)alloc((size_t)BATCH * CC * 4);

    const int nw = CC * CC;
    const int nm = CC * KMIX;
    k_cvt_f16<<<(nw + 255) / 256, 256, 0, stream>>>(Wh,       Wh16, nw);
    k_cvt_f16<<<(nw + 255) / 256, 256, 0, stream>>>(Ww_,      Ww16, nw);
    k_cvt_f16<<<(nw + 255) / 256, 256, 0, stream>>>(Ws,       Ws16, nw);
    k_cvt_f16<<<(nw + 255) / 256, 256, 0, stream>>>(proj_W,   Wp16, nw);
    k_cvt_f16<<<(nm + 255) / 256, 256, 0, stream>>>(mixall_W, Wm16, nm);

    dim3 gg(MROWS / 256, CC / 64);   // 98 x 12
    k_gemm_s1<<<gg, 256, 0, stream>>>(x, Wh16, bh, h16);
    k_gemm_s1<<<gg, 256, 0, stream>>>(x, Ww16, bw, w16);
    k_gemm_s1<<<gg, 256, 0, stream>>>(x, Ws16, bs, s16);

    k_pool_h<<<BATCH * HH, 256, 0, stream>>>(h16, h2v);
    k_pool_w<<<BATCH * WW, 256, 0, stream>>>(w16, w2v);
    k_pool_s<<<BATCH,      256, 0, stream>>>(s16, s1v);

    k_gates<<<BATCH, 256, 0, stream>>>(h2v, w2v, s1v,
                                       convH_w, convH_b, convH2_k, convH2_b,
                                       mixhw_k, mixhw_b, mixhw2_k, mixhw2_b,
                                       mixhws_k, mixhws_b, mixhws2_k, mixhws2_b,
                                       linH_W, linH_b, linW_W, linW_b,
                                       g_h, g_hw, g_c);

    k_gemm_mix<<<gg, 256, 0, stream>>>(h16, w16, s16, Wm16, mixall_b,
                                       g_h, g_hw, g_c, mix16);

    k_gemm_proj<<<gg, 256, 0, stream>>>(mix16, Wp16, proj_b, x, gate_w, out);

    (void)in_sizes; (void)n_in; (void)out_size; (void)ws_size;
}
